// GCN_36412732735797
// MI455X (gfx1250) — matrix operand order repaired and checked by reading, not compiled, not run on hardware
//
#include <hip/hip_runtime.h>
#include <hip/hip_bf16.h>

// Problem constants (match reference)
#define NNODES 100000
#define NEDGES 1600000
#define NGRAPH 100
#define DIN    3
#define HID    128
#define DOUT   10
#define MTILES (NNODES / 16)   // 6250 exactly
#define MPAIRS (MTILES / 2)    // 3125: each wave handles two 16-row tiles
#define BPAD   136             // padded LDS row stride in halves (272B = 68 dwords)

typedef _Float16 v4h  __attribute__((ext_vector_type(4)));
typedef _Float16 v8h  __attribute__((ext_vector_type(8)));
typedef _Float16 v16h __attribute__((ext_vector_type(16)));
typedef float    v8f  __attribute__((ext_vector_type(8)));

// ---------------------------------------------------------------------------
// Utility kernels
// ---------------------------------------------------------------------------
__global__ void zero_f32(float* __restrict__ p, int n) {
    int i = blockIdx.x * blockDim.x + threadIdx.x;
    if (i < n) p[i] = 0.0f;
}

__global__ void cvt_f32_to_f16(const float* __restrict__ s, _Float16* __restrict__ d, int n) {
    int i = blockIdx.x * blockDim.x + threadIdx.x;
    if (i < n) d[i] = (_Float16)s[i];
}

// deg[dst] += 1 per edge (self-loop +1 added in finalize)
__global__ void deg_count(const int* __restrict__ dst, float* __restrict__ deg, int e) {
    int i = blockIdx.x * blockDim.x + threadIdx.x;
    if (i < e) atomicAdd(&deg[dst[i]], 1.0f);
}

// deg -> dinv_sqrt = rsqrt(deg + 1)
__global__ void deg_finalize(float* __restrict__ deg, int n) {
    int i = blockIdx.x * blockDim.x + threadIdx.x;
    if (i < n) deg[i] = rsqrtf(deg[i] + 1.0f);
}

// norm[e] = dinv[src] * dinv[dst]
__global__ void edge_norm(const float* __restrict__ dinv, const int* __restrict__ src,
                          const int* __restrict__ dst, float* __restrict__ nrm, int e) {
    int i = blockIdx.x * blockDim.x + threadIdx.x;
    if (i < e) nrm[i] = dinv[src[i]] * dinv[dst[i]];
}

// ---------------------------------------------------------------------------
// Layer 1 dense part: h = x @ W1  (K = 3, pure VALU; wave-per-node, 4 feats/lane)
// ---------------------------------------------------------------------------
__global__ void xw1_kernel(const float* __restrict__ x, const float* __restrict__ W1,
                           float* __restrict__ h, int n) {
    int t = blockIdx.x * blockDim.x + threadIdx.x;
    int i = __builtin_amdgcn_readfirstlane(t >> 5);   // wave-uniform node id
    if (i >= n) return;
    int f = (t & 31) * 4;
    float x0 = x[(size_t)i * 3 + 0];
    float x1 = x[(size_t)i * 3 + 1];
    float x2 = x[(size_t)i * 3 + 2];
    float4 w0 = *(const float4*)(W1 + 0 * HID + f);
    float4 w1 = *(const float4*)(W1 + 1 * HID + f);
    float4 w2 = *(const float4*)(W1 + 2 * HID + f);
    float4 r;
    r.x = fmaf(x0, w0.x, fmaf(x1, w1.x, x2 * w2.x));
    r.y = fmaf(x0, w0.y, fmaf(x1, w1.y, x2 * w2.y));
    r.z = fmaf(x0, w0.z, fmaf(x1, w1.z, x2 * w2.z));
    r.w = fmaf(x0, w0.w, fmaf(x1, w1.w, x2 * w2.w));
    *(float4*)(h + (size_t)i * HID + f) = r;
}

// ---------------------------------------------------------------------------
// Edge scatter: agg[dst] += norm[e] * h[src].  Wave per edge, float4 per lane:
// coalesced 512B gather; scatter via global_atomic_add_f32 (L2-resident).
// Edge index is wave-uniform -> scalarize src/dst/norm loads off the VMEM path.
// ---------------------------------------------------------------------------
__global__ void edge_msg(const float* __restrict__ h, const float* __restrict__ nrm,
                         const int* __restrict__ src, const int* __restrict__ dst,
                         float* __restrict__ agg, int e) {
    int t = blockIdx.x * blockDim.x + threadIdx.x;
    int eidx = __builtin_amdgcn_readfirstlane(t >> 5);
    if (eidx >= e) return;
    int lane = t & 31;
    int s = src[eidx];
    int d = dst[eidx];
    float w = nrm[eidx];
    float4 hv = *(const float4*)(h + (size_t)s * HID + lane * 4);
    float* ap = agg + (size_t)d * HID + lane * 4;
    atomicAdd(ap + 0, hv.x * w);
    atomicAdd(ap + 1, hv.y * w);
    atomicAdd(ap + 2, hv.z * w);
    atomicAdd(ap + 3, hv.w * w);
}

// ---------------------------------------------------------------------------
// Epilogue 1: relu(agg + h/deg + b1) -> f16 (feeds WMMA GEMM)
// ---------------------------------------------------------------------------
__global__ void epilogue1(const float* __restrict__ agg, const float* __restrict__ h,
                          const float* __restrict__ dinv, const float* __restrict__ b,
                          _Float16* __restrict__ out16, int n) {
    int t = blockIdx.x * blockDim.x + threadIdx.x;
    int i = __builtin_amdgcn_readfirstlane(t >> 5);
    if (i >= n) return;
    int f = (t & 31) * 4;
    float ds = dinv[i];
    float inv = ds * ds;           // 1/deg = dinv_sqrt^2
    float4 a  = *(const float4*)(agg + (size_t)i * HID + f);
    float4 hv = *(const float4*)(h + (size_t)i * HID + f);
    float4 bv = *(const float4*)(b + f);
    v4h o;
    o[0] = (_Float16)fmaxf(fmaf(hv.x, inv, a.x) + bv.x, 0.0f);
    o[1] = (_Float16)fmaxf(fmaf(hv.y, inv, a.y) + bv.y, 0.0f);
    o[2] = (_Float16)fmaxf(fmaf(hv.z, inv, a.z) + bv.z, 0.0f);
    o[3] = (_Float16)fmaxf(fmaf(hv.w, inv, a.w) + bv.w, 0.0f);
    *(v4h*)(out16 + (size_t)i * HID + f) = o;
}

// ---------------------------------------------------------------------------
// Layer 2 GEMM: C[100000,128] = A_f16[100000,128] @ B_f16[128,128], f32 accum.
// W2 panel (32KB) staged once per block into LDS via global_load_async_to_lds,
// padded rows (272B) to spread LDS banks. One wave per PAIR of 16-row M-tiles;
// A panel kept in registers (4x v16h); 8 N-tiles x 4 K-steps = 32 WMMAs/tile.
// VGPR layouts per CDNA5 ISA 7.12.2.
// ---------------------------------------------------------------------------
__global__ void __launch_bounds__(256)
gemm_wmma_f16(const _Float16* __restrict__ A, const _Float16* __restrict__ B,
              float* __restrict__ C, int mpairs) {
    __shared__ _Float16 sB[128 * BPAD];

    const int tid = threadIdx.x;

    // ---- async stage of B into LDS: thread t copies row (t>>1), half (t&1) ----
    {
        const int row = tid >> 1;                       // 0..127
        const int seg = tid & 1;                        // 0/1 -> 64 halves each
        const _Float16* g = B + (size_t)row * HID + seg * 64;
        unsigned long long gbase = (unsigned long long)(size_t)g;
        // generic LDS pointer: addr[31:0] is the LDS byte offset (ISA 10.2)
        unsigned lbase = (unsigned)(size_t)(sB + (size_t)row * BPAD + seg * 64);
#pragma unroll
        for (int k = 0; k < 8; ++k) {                   // 8 x 16B = 128B
            unsigned loff = lbase + k * 16;
            unsigned long long ga = gbase + (unsigned long long)(k * 16);
            asm volatile("global_load_async_to_lds_b128 %0, %1, off"
                         :: "v"(loff), "v"(ga) : "memory");
        }
        asm volatile("s_wait_asynccnt 0x0" ::: "memory");
    }
    __syncthreads();

    const int wid  = blockIdx.x * (blockDim.x >> 5) + (tid >> 5);
    const int lane = tid & 31;
    if (wid < mpairs) {                     // wave-uniform: EXEC all-1s inside
        const int row = lane & 15;
        const int hi  = lane >> 4;          // lane half selects K sub-block

#pragma unroll
        for (int sub = 0; sub < 2; ++sub) {
            const int m0 = (wid * 2 + sub) << 4;

            // A: 16x32 f16 tile per K-step. Lane (row, half): halves 0..7 =
            // K[kb..kb+7], halves 8..15 = K[kb+16..kb+23], kb = 32*kk + 8*hi.
            const _Float16* arow = A + (size_t)(m0 + row) * HID;
            v16h a[4];
#pragma unroll
            for (int kk = 0; kk < 4; ++kk) {
                const int kb = kk * 32 + hi * 8;
                union { v16h v; v8h h[2]; } u;
                u.h[0] = *(const v8h*)(arow + kb);
                u.h[1] = *(const v8h*)(arow + kb + 16);
                a[kk] = u.v;
            }

#pragma unroll
            for (int nt = 0; nt < 8; ++nt) {
                v8f acc = {};
#pragma unroll
                for (int kk = 0; kk < 4; ++kk) {
                    // B: 32x16 f16 tile; lane = N column, halves 0..7 = K[kb..kb+7], 8..15 = K[kb+16..kb+23] (LDS).
                    const _Float16* bp = sB + (size_t)(kk * 32 + hi * 8) * BPAD + nt * 16 + row;
                    union { v16h v; _Float16 h[16]; } ub;
#pragma unroll
                    for (int i = 0; i < 8; ++i) {
                        ub.h[i]     = bp[(size_t)i * BPAD];
                        ub.h[8 + i] = bp[(size_t)(16 + i) * BPAD];
                    }
                    acc = __builtin_amdgcn_wmma_f32_16x16x32_f16(
                        false, a[kk], false, ub.v, (short)0, acc, false, false);
                }
                // C: VGPR r -> M = r + 8*hi, N = lane&15
                float* cp = C + (size_t)(m0 + hi * 8) * HID + nt * 16 + row;
#pragma unroll
                for (int r = 0; r < 8; ++r)
                    cp[(size_t)r * HID] = acc[r];
            }
        }
    }
}

// ---------------------------------------------------------------------------
// Epilogue 2 fused with mean-pool accumulation:
// v = relu(agg + h/deg + b2); poolsum[batch[i]] += v  (atomic)
// ---------------------------------------------------------------------------
__global__ void epilogue2_pool(const float* __restrict__ agg, const float* __restrict__ h,
                               const float* __restrict__ dinv, const float* __restrict__ b,
                               const int* __restrict__ batch, float* __restrict__ psum, int n) {
    int t = blockIdx.x * blockDim.x + threadIdx.x;
    int i = __builtin_amdgcn_readfirstlane(t >> 5);
    if (i >= n) return;
    int f = (t & 31) * 4;
    float ds = dinv[i];
    float inv = ds * ds;
    float4 a  = *(const float4*)(agg + (size_t)i * HID + f);
    float4 hv = *(const float4*)(h + (size_t)i * HID + f);
    float4 bv = *(const float4*)(b + f);
    float v0 = fmaxf(fmaf(hv.x, inv, a.x) + bv.x, 0.0f);
    float v1 = fmaxf(fmaf(hv.y, inv, a.y) + bv.y, 0.0f);
    float v2 = fmaxf(fmaf(hv.z, inv, a.z) + bv.z, 0.0f);
    float v3 = fmaxf(fmaf(hv.w, inv, a.w) + bv.w, 0.0f);
    float* pp = psum + (size_t)batch[i] * HID + f;
    atomicAdd(pp + 0, v0);
    atomicAdd(pp + 1, v1);
    atomicAdd(pp + 2, v2);
    atomicAdd(pp + 3, v3);
}

// Per-block LDS histogram, then one flush: 100 global atomics per block
// instead of 256 highly-contended ones.
__global__ void count_nodes(const int* __restrict__ batch, float* __restrict__ cnt, int n) {
    __shared__ float hc[NGRAPH];
    for (int j = threadIdx.x; j < NGRAPH; j += blockDim.x) hc[j] = 0.0f;
    __syncthreads();
    int i = blockIdx.x * blockDim.x + threadIdx.x;
    if (i < n) atomicAdd(&hc[batch[i]], 1.0f);
    __syncthreads();
    for (int j = threadIdx.x; j < NGRAPH; j += blockDim.x)
        if (hc[j] != 0.0f) atomicAdd(&cnt[j], hc[j]);
}

// ---------------------------------------------------------------------------
// Final: out[g,o] = (poolsum[g]/max(cnt,1)) . Wl[:,o] + bl[o]
// ---------------------------------------------------------------------------
__global__ void final_linear(const float* __restrict__ psum, const float* __restrict__ cnt,
                             const float* __restrict__ Wl, const float* __restrict__ bl,
                             float* __restrict__ out) {
    int t = blockIdx.x * blockDim.x + threadIdx.x;
    if (t >= NGRAPH * DOUT) return;
    int g = t / DOUT;
    int o = t % DOUT;
    float c = fmaxf(cnt[g], 1.0f);
    float s = 0.0f;
#pragma unroll 8
    for (int f = 0; f < HID; ++f)
        s = fmaf(psum[(size_t)g * HID + f], Wl[(size_t)f * DOUT + o], s);
    out[t] = s / c + bl[o];
}

// ---------------------------------------------------------------------------
// Host launcher
// ---------------------------------------------------------------------------
extern "C" void kernel_launch(void* const* d_in, const int* in_sizes, int n_in,
                              void* d_out, int out_size, void* d_ws, size_t ws_size,
                              hipStream_t stream) {
    (void)in_sizes; (void)n_in; (void)out_size; (void)ws_size;

    const float* x    = (const float*)d_in[0];
    const int*   ei   = (const int*)d_in[1];
    const int*   src  = ei;
    const int*   dst  = ei + NEDGES;
    const int*   batch= (const int*)d_in[2];
    const float* W1   = (const float*)d_in[3];
    const float* b1   = (const float*)d_in[4];
    const float* W2   = (const float*)d_in[5];
    const float* b2   = (const float*)d_in[6];
    const float* Wl   = (const float*)d_in[7];
    const float* bl   = (const float*)d_in[8];
    float* out = (float*)d_out;

    // Workspace carve-out (256B aligned): ~135 MB total
    char* w = (char*)d_ws;
    size_t off = 0;
    auto alloc = [&](size_t bytes) -> void* {
        void* p = w + off;
        off = (off + bytes + 255) & ~(size_t)255;
        return p;
    };
    float*     deg  = (float*)alloc((size_t)NNODES * 4);          // then dinv_sqrt
    float*     nrm  = (float*)alloc((size_t)NEDGES * 4);
    float*     hbuf = (float*)alloc((size_t)NNODES * HID * 4);    // pre-message h (both layers)
    float*     agg  = (float*)alloc((size_t)NNODES * HID * 4);    // scatter target (both layers)
    _Float16*  h16  = (_Float16*)alloc((size_t)NNODES * HID * 2); // layer-1 relu output (f16)
    _Float16*  w2h  = (_Float16*)alloc((size_t)HID * HID * 2);
    float*     psum = (float*)alloc((size_t)NGRAPH * HID * 4);
    float*     cnt  = (float*)alloc((size_t)NGRAPH * 4);

    const int B = 256;
    auto blocks = [](long long n, int b) { return (unsigned)((n + b - 1) / b); };

    // --- init accumulators ---
    zero_f32<<<blocks(NNODES, B), B, 0, stream>>>(deg, NNODES);
    zero_f32<<<blocks((long long)NNODES * HID, B), B, 0, stream>>>(agg, NNODES * HID);
    zero_f32<<<blocks(NGRAPH * HID, B), B, 0, stream>>>(psum, NGRAPH * HID);
    zero_f32<<<blocks(NGRAPH, B), B, 0, stream>>>(cnt, NGRAPH);

    // --- degrees and edge norms ---
    deg_count<<<blocks(NEDGES, B), B, 0, stream>>>(dst, deg, NEDGES);
    deg_finalize<<<blocks(NNODES, B), B, 0, stream>>>(deg, NNODES);   // deg := rsqrt(deg+1)
    edge_norm<<<blocks(NEDGES, B), B, 0, stream>>>(deg, src, dst, nrm, NEDGES);

    // --- layer 1 ---
    xw1_kernel<<<blocks((long long)NNODES * 32, B), B, 0, stream>>>(x, W1, hbuf, NNODES);
    edge_msg<<<blocks((long long)NEDGES * 32, B), B, 0, stream>>>(hbuf, nrm, src, dst, agg, NEDGES);
    epilogue1<<<blocks((long long)NNODES * 32, B), B, 0, stream>>>(agg, hbuf, deg, b1, h16, NNODES);

    // --- layer 2 ---
    cvt_f32_to_f16<<<blocks(HID * HID, B), B, 0, stream>>>(W2, w2h, HID * HID);
    zero_f32<<<blocks((long long)NNODES * HID, B), B, 0, stream>>>(agg, NNODES * HID);
    gemm_wmma_f16<<<blocks(MPAIRS, 8), B, 0, stream>>>(h16, w2h, hbuf, MPAIRS);
    edge_msg<<<blocks((long long)NEDGES * 32, B), B, 0, stream>>>(hbuf, nrm, src, dst, agg, NEDGES);

    // --- pooling (fused into epilogue 2) + readout ---
    count_nodes<<<blocks(NNODES, B), B, 0, stream>>>(batch, cnt, NNODES);
    epilogue2_pool<<<blocks((long long)NNODES * 32, B), B, 0, stream>>>(agg, hbuf, deg, b2, batch, psum, NNODES);
    final_linear<<<blocks(NGRAPH * DOUT, B), B, 0, stream>>>(psum, cnt, Wl, bl, out);
}
